// Model_38869454028874
// MI455X (gfx1250) — matrix operand order repaired and checked by reading, not compiled, not run on hardware
//
#include <hip/hip_runtime.h>
#include <hip/hip_bf16.h>

typedef __attribute__((ext_vector_type(16))) _Float16 v16h;
typedef __attribute__((ext_vector_type(8)))  float    v8f;

#define M_ROWS 8000
#define NTILES 500   // 8000 / 16
#define USPLIT 20    // split of u-tile range in the Gram kernel (500/20 = 25 tiles per wave)

// ---------------------------------------------------------------------------
// Zero the accumulator region of the workspace (srr_pos, srr_neg, rowsum, Sp/Sn)
// ---------------------------------------------------------------------------
__global__ void k_zero(float* __restrict__ ws, int n) {
    int i = blockIdx.x * blockDim.x + threadIdx.x;
    if (i < n) ws[i] = 0.f;
}

// ---------------------------------------------------------------------------
// fc = normalize([c, 0]);  also keep fc_raw
// ---------------------------------------------------------------------------
__global__ void k_fc(const float* __restrict__ c, float* __restrict__ fc,
                     float* __restrict__ fcraw) {
    __shared__ float sh[64];
    int t = threadIdx.x;                       // 64 threads
    float x = (t < 63) ? c[t] : 0.f;
    sh[t] = x * x;
    __syncthreads();
    float ss = 0.f;
    for (int k = 0; k < 64; ++k) ss += sh[k];
    float inv = rsqrtf(ss);
    fcraw[t] = x;
    fc[t]    = x * inv;
}

// ---------------------------------------------------------------------------
// Per-row features: Fr_raw = [A[v,:], b[v]]; Fr = Fr_raw / ||Fr_raw||
//  - Fr16: normalized rows in f16 (for WMMA Gram)
//  - mu0 : raw rows in f16 (mu_r0 = Fr_raw^T, stored transposed as [v][p])
//  - wrc[v] = Fr[v].fc ;  Sp += relu(wrc), Sn += relu(-wrc)
// One wave per row; lane l handles elements l and l+32.
// ---------------------------------------------------------------------------
__global__ void k_rows(const float* __restrict__ A, const float* __restrict__ b,
                       const float* __restrict__ fc, float* __restrict__ wrc,
                       float* __restrict__ scal,
                       _Float16* __restrict__ Fr16, _Float16* __restrict__ mu0) {
    const int lane = threadIdx.x & 31;
    const int wv   = threadIdx.x >> 5;         // 0..7
    const int v    = blockIdx.x * 8 + wv;
    const int e0 = lane, e1 = lane + 32;
    float x0 = A[(size_t)v * 63 + e0];                       // e0 <= 31 < 63
    float x1 = (e1 < 63) ? A[(size_t)v * 63 + e1] : b[v];    // e1 == 63 -> b
    float ss = x0 * x0 + x1 * x1;
#pragma unroll
    for (int off = 16; off >= 1; off >>= 1) ss += __shfl_xor(ss, off, 32);
    float inv = rsqrtf(ss);
    float f0 = x0 * inv, f1 = x1 * inv;
    float wd = f0 * fc[e0] + f1 * fc[e1];
#pragma unroll
    for (int off = 16; off >= 1; off >>= 1) wd += __shfl_xor(wd, off, 32);
    if (lane == 0) {
        wrc[v] = wd;
        atomicAdd(&scal[0], fmaxf(wd, 0.f));
        atomicAdd(&scal[1], fmaxf(-wd, 0.f));
    }
    Fr16[(size_t)v * 64 + e0] = (_Float16)f0;
    Fr16[(size_t)v * 64 + e1] = (_Float16)f1;
    mu0 [(size_t)v * 64 + e0] = (_Float16)x0;
    mu0 [(size_t)v * 64 + e1] = (_Float16)x1;
}

// ---------------------------------------------------------------------------
// Gram reduction: srr_pos[v] = sum_{u!=v} relu(Fr[v].Fr[u]); srr_neg likewise.
// One wave owns TWO 16-row v-tiles (32 rows): each B-tile load feeds 4 WMMAs.
// The hot loop is completely branch-free (diagonal included); the diagonal
// contribution d_v = |q(Fr[v])|^2 is recomputed exactly from the resident
// A-fragments and subtracted once after the loop.
// u-tiles split USPLIT ways (grid.y).
// ---------------------------------------------------------------------------
__global__ void k_gram(const _Float16* __restrict__ Fr16,
                       float* __restrict__ srr_pos, float* __restrict__ srr_neg) {
    const int lane = threadIdx.x & 31;
    const int wv   = threadIdx.x >> 5;         // 2 waves / block (64 threads)
    const int pt   = blockIdx.x * 2 + wv;      // v-pair index: 0..249
    const int v0   = pt * 32;                  // 32 rows per wave
    const int m    = lane & 15;
    const int hi   = lane >> 4;
    const int t0   = pt * 2;                   // this wave's two v-tile indices
    const int t1   = pt * 2 + 1;

    // A fragments (loop-invariant): rows v0+m (tile 0) and v0+16+m (tile 1)
    // A layout (16-bit, 16x32): slot i -> K = i + 8*(i>=8) + 8*hi (+kbase)
    const _Float16* arow0 = Fr16 + (size_t)(v0 + m) * 64;
    const _Float16* arow1 = Fr16 + (size_t)(v0 + 16 + m) * 64;
    v16h a00, a01, a10, a11;
#pragma unroll
    for (int i = 0; i < 8; ++i) {
        a00[i]     = arow0[ 0 + 8 * hi + i];
        a00[i + 8] = arow0[16 + 8 * hi + i];
        a01[i]     = arow0[32 + 8 * hi + i];
        a01[i + 8] = arow0[48 + 8 * hi + i];
        a10[i]     = arow1[ 0 + 8 * hi + i];
        a10[i + 8] = arow1[16 + 8 * hi + i];
        a11[i]     = arow1[32 + 8 * hi + i];
        a11[i + 8] = arow1[48 + 8 * hi + i];
    }

    float pos[2][8], neg[2][8];
#pragma unroll
    for (int ct = 0; ct < 2; ++ct)
#pragma unroll
        for (int r = 0; r < 8; ++r) { pos[ct][r] = 0.f; neg[ct][r] = 0.f; }

    const int uPer = NTILES / USPLIT;          // 25
    const int uBeg = blockIdx.y * uPer;
    const int uEnd = uBeg + uPer;
    for (int ut = uBeg; ut < uEnd; ++ut) {
        const int u0 = ut * 16;
        // B layout (32x16): slot j -> K = j + 8*(j>=8) + 8*hi (+kbase), N = m
        const _Float16* brow = Fr16 + (size_t)(u0 + m) * 64;
        v16h b0, b1;
#pragma unroll
        for (int j = 0; j < 8; ++j) {
            b0[j]     = brow[ 0 + 8 * hi + j];
            b0[j + 8] = brow[16 + 8 * hi + j];
            b1[j]     = brow[32 + 8 * hi + j];
            b1[j + 8] = brow[48 + 8 * hi + j];
        }
        v8f c0 = {}, c1 = {};
        c0 = __builtin_amdgcn_wmma_f32_16x16x32_f16(false, a00, false, b0,
                                                    (short)0, c0, false, false);
        c1 = __builtin_amdgcn_wmma_f32_16x16x32_f16(false, a10, false, b0,
                                                    (short)0, c1, false, false);
        c0 = __builtin_amdgcn_wmma_f32_16x16x32_f16(false, a01, false, b1,
                                                    (short)0, c0, false, false);
        c1 = __builtin_amdgcn_wmma_f32_16x16x32_f16(false, a11, false, b1,
                                                    (short)0, c1, false, false);
        // branch-free relu accumulation (dual-issue friendly)
#pragma unroll
        for (int r = 0; r < 8; ++r) {
            float x0 = c0[r], x1 = c1[r];
            pos[0][r] += fmaxf(x0, 0.f);
            neg[0][r] += fmaxf(-x0, 0.f);
            pos[1][r] += fmaxf(x1, 0.f);
            neg[1][r] += fmaxf(-x1, 0.f);
        }
    }

    // --- one-time diagonal correction -------------------------------------
    // d for row v0+m (tile 0) / v0+16+m (tile 1): each lane's 32 A halves
    // cover half the K range; shfl_xor(16) completes the sum.
    float sq0 = 0.f, sq1 = 0.f;
#pragma unroll
    for (int i = 0; i < 16; ++i) {
        float u00 = (float)a00[i], u01 = (float)a01[i];
        float u10 = (float)a10[i], u11 = (float)a11[i];
        sq0 += u00 * u00 + u01 * u01;
        sq1 += u10 * u10 + u11 * u11;
    }
    const float d0 = sq0 + __shfl_xor(sq0, 16, 32);   // row v0 + m
    const float d1 = sq1 + __shfl_xor(sq1, 16, 32);   // row v0 + 16 + m
    // Diagonal of tile t lands at slot r of the lane where m == r + 8*hi.
    const bool in0 = (t0 >= uBeg) && (t0 < uEnd);
    const bool in1 = (t1 >= uBeg) && (t1 < uEnd);
#pragma unroll
    for (int r = 0; r < 8; ++r) {
        const bool sel = (m == r + 8 * hi);
        pos[0][r] -= (sel && in0) ? fmaxf(d0, 0.f) : 0.f;
        neg[0][r] -= (sel && in0) ? fmaxf(-d0, 0.f) : 0.f;
        pos[1][r] -= (sel && in1) ? fmaxf(d1, 0.f) : 0.f;
        neg[1][r] -= (sel && in1) ? fmaxf(-d1, 0.f) : 0.f;
    }

    // reduce over the 16 lanes of each half (N dimension), then atomically add
#pragma unroll
    for (int ct = 0; ct < 2; ++ct) {
#pragma unroll
        for (int r = 0; r < 8; ++r) {
#pragma unroll
            for (int off = 8; off >= 1; off >>= 1) {
                pos[ct][r] += __shfl_xor(pos[ct][r], off, 32);
                neg[ct][r] += __shfl_xor(neg[ct][r], off, 32);
            }
        }
    }
    if (m == 0) {
#pragma unroll
        for (int ct = 0; ct < 2; ++ct)
#pragma unroll
            for (int r = 0; r < 8; ++r) {
                const int row = v0 + 16 * ct + r + 8 * hi;
                atomicAdd(&srr_pos[row], pos[ct][r]);
                atomicAdd(&srr_neg[row], neg[ct][r]);
            }
    }
}

// ---------------------------------------------------------------------------
// Scalar prep (1 block, 64 threads): fz, t1r/t1c, edge rank-4 vectors, edge_c,
// vconst0 = t1r + th2rc@fc_raw, mu_c0 = fc_raw, th2rr -> f16.
// ---------------------------------------------------------------------------
__global__ void k_prep(const float* __restrict__ A, const float* __restrict__ b,
                       const int* __restrict__ zp,
                       const float* __restrict__ th1r, const float* __restrict__ th1c,
                       const float* __restrict__ th3rr, const float* __restrict__ th3rc,
                       const float* __restrict__ th3cr,
                       const float* __restrict__ th4rr, const float* __restrict__ th4rc,
                       const float* __restrict__ th4cr,
                       const float* __restrict__ th2rc, const float* __restrict__ th2rr,
                       const float* __restrict__ scal, const float* __restrict__ fcraw,
                       float* __restrict__ t1r, float* __restrict__ t1c,
                       float* __restrict__ evec, float* __restrict__ edge_c,
                       float* __restrict__ vconst, float* __restrict__ mu_c,
                       _Float16* __restrict__ th2rr16) {
    __shared__ float fz[64];
    __shared__ float scr[64];
    int t = threadIdx.x;                       // 64 threads
    int z = *zp;
    float x = (t < 63) ? A[(size_t)z * 63 + t] : b[z];
    fz[t] = x;
    __syncthreads();
    float ss = 0.f;
    for (int k = 0; k < 64; ++k) ss += fz[k] * fz[k];
    float inv = rsqrtf(ss);

    float a1 = 0.f, a2 = 0.f;
    for (int k = 0; k < 64; ++k) {
        float f = fz[k] * inv;
        a1 += th1r[t * 64 + k] * f;
        a2 += th1c[t * 64 + k] * f;
    }
    t1r[t] = a1;
    t1c[t] = a2;

    float e1 = 0.f, e2 = 0.f, e3 = 0.f, e4 = 0.f;
    for (int k = 0; k < 64; ++k) {
        e1 += th3rr[t * 64 + k] * fmaxf(th4rr[k], 0.f);
        e2 += th3rr[t * 64 + k] * fmaxf(-th4rr[k], 0.f);
        e3 += th3rc[t * 64 + k] * fmaxf(th4rc[k], 0.f);
        e4 += th3rc[t * 64 + k] * fmaxf(-th4rc[k], 0.f);
    }
    evec[t] = e1; evec[64 + t] = e2; evec[128 + t] = e3; evec[192 + t] = e4;

    float Sp = scal[0], Sn = scal[1];
    scr[t] = fmaxf(th4cr[t], 0.f) * Sp + fmaxf(-th4cr[t], 0.f) * Sn;
    __syncthreads();
    float ec = 0.f;
    for (int k = 0; k < 64; ++k) ec += th3cr[t * 64 + k] * scr[k];
    edge_c[t] = ec;

    float vc = a1;
    for (int k = 0; k < 64; ++k) vc += th2rc[t * 64 + k] * fcraw[k];
    vconst[t] = vc;
    mu_c[t] = fcraw[t];

    for (int k = 0; k < 64; ++k)
        th2rr16[t * 64 + k] = (_Float16)th2rr[t * 64 + k];
}

// ---------------------------------------------------------------------------
// Row update: new_r[:,v] = relu(vconst + th2rr@mu_r[:,v] + e1*sp + e2*sn +
//                               e3*relu(wrc) + e4*relu(-wrc))
// One wave per 16-column tile; 4 p-tiles x 2 k-steps = 8 WMMAs per wave.
// mu stored transposed [v][p] (f16). Accumulates rowsum (sum over v) via atomics.
// ---------------------------------------------------------------------------
__global__ void k_row(const _Float16* __restrict__ muIn, _Float16* __restrict__ muOut,
                      const _Float16* __restrict__ th2rr16,
                      const float* __restrict__ vconst, const float* __restrict__ evec,
                      const float* __restrict__ srr_pos, const float* __restrict__ srr_neg,
                      const float* __restrict__ wrc, float* __restrict__ rowsum) {
    const int lane  = threadIdx.x & 31;
    const int wv    = threadIdx.x >> 5;
    const int vtile = blockIdx.x * 4 + wv;
    const int v0    = vtile * 16;
    const int m     = lane & 15;
    const int hi    = lane >> 4;
    const int v     = v0 + m;

    const _Float16* brow = muIn + (size_t)v * 64;
    v16h b0, b1;
#pragma unroll
    for (int j = 0; j < 8; ++j) {
        b0[j]     = brow[ 0 + 8 * hi + j];
        b0[j + 8] = brow[16 + 8 * hi + j];
        b1[j]     = brow[32 + 8 * hi + j];
        b1[j + 8] = brow[48 + 8 * hi + j];
    }
    const float sp = srr_pos[v], sn = srr_neg[v];
    const float w  = wrc[v];
    const float rp = fmaxf(w, 0.f), rn = fmaxf(-w, 0.f);

    float sums[4][8];
#pragma unroll
    for (int t = 0; t < 4; ++t) {
        const _Float16* arow = th2rr16 + (size_t)(16 * t + m) * 64;
        v16h a0, a1;
#pragma unroll
        for (int i = 0; i < 8; ++i) {
            a0[i]     = arow[ 0 + 8 * hi + i];
            a0[i + 8] = arow[16 + 8 * hi + i];
            a1[i]     = arow[32 + 8 * hi + i];
            a1[i + 8] = arow[48 + 8 * hi + i];
        }
        v8f cc = {};
        cc = __builtin_amdgcn_wmma_f32_16x16x32_f16(false, a0, false, b0,
                                                    (short)0, cc, false, false);
        cc = __builtin_amdgcn_wmma_f32_16x16x32_f16(false, a1, false, b1,
                                                    (short)0, cc, false, false);
#pragma unroll
        for (int r = 0; r < 8; ++r) {
            const int p = 16 * t + r + 8 * hi;
            float val = cc[r] + vconst[p] + evec[p] * sp + evec[64 + p] * sn +
                        evec[128 + p] * rp + evec[192 + p] * rn;
            sums[t][r] = fmaxf(val, 0.f);
        }
    }
    // store new mu (transposed, f16), packing half pairs into dwords
#pragma unroll
    for (int t = 0; t < 4; ++t) {
#pragma unroll
        for (int r = 0; r < 8; r += 2) {
            union { _Float16 h[2]; unsigned u; } pk;
            pk.h[0] = (_Float16)sums[t][r];
            pk.h[1] = (_Float16)sums[t][r + 1];
            *reinterpret_cast<unsigned*>(muOut + (size_t)v * 64 + 16 * t + 8 * hi + r) = pk.u;
        }
    }
    // rowsum accumulation: reduce over the 16 columns held by each half
#pragma unroll
    for (int t = 0; t < 4; ++t) {
#pragma unroll
        for (int r = 0; r < 8; ++r) {
            float s = sums[t][r];
#pragma unroll
            for (int off = 8; off >= 1; off >>= 1) s += __shfl_xor(s, off, 32);
            if (m == 0) atomicAdd(&rowsum[16 * t + r + 8 * hi], s);
        }
    }
}

// ---------------------------------------------------------------------------
// Column update (1 block, 64 threads): mu_c = relu(t1c + th2cr@rowsum + edge_c)
// Also prepares vconst for next step and re-zeroes rowsum (saving a copy).
// ---------------------------------------------------------------------------
__global__ void k_col(const float* __restrict__ t1c, const float* __restrict__ t1r,
                      const float* __restrict__ th2cr, const float* __restrict__ th2rc,
                      const float* __restrict__ edge_c,
                      float* __restrict__ rowsum, float* __restrict__ mu_c,
                      float* __restrict__ vconst, float* __restrict__ last_rs) {
    __shared__ float rs[64], mc[64];
    int t = threadIdx.x;
    rs[t] = rowsum[t];
    __syncthreads();
    float acc = t1c[t] + edge_c[t];
    for (int k = 0; k < 64; ++k) acc += th2cr[t * 64 + k] * rs[k];
    float nc = fmaxf(acc, 0.f);
    mc[t] = nc;
    __syncthreads();
    mu_c[t]    = nc;
    last_rs[t] = rs[t];
    rowsum[t]  = 0.f;
    float vc = t1r[t];
    for (int k = 0; k < 64; ++k) vc += th2rc[t * 64 + k] * mc[k];
    vconst[t] = vc;
}

// ---------------------------------------------------------------------------
// Head (1 block, 64 threads): term6/term7 -> sigmoid -> W8 -> out (1,2)
// ---------------------------------------------------------------------------
__global__ void k_final(const float* __restrict__ th6r, const float* __restrict__ th6c,
                        const float* __restrict__ th7, const float* __restrict__ W8,
                        const float* __restrict__ b8, const int* __restrict__ zp,
                        const _Float16* __restrict__ muF,
                        const float* __restrict__ last_rs, const float* __restrict__ mu_c,
                        float* __restrict__ out) {
    __shared__ float ls[64], mc[64], mz[64], feat[128];
    int t = threadIdx.x;
    int z = *zp;
    ls[t] = last_rs[t];
    mc[t] = mu_c[t];
    mz[t] = (float)muF[(size_t)z * 64 + t];
    __syncthreads();
    float t6 = 0.f, t7 = 0.f;
    for (int k = 0; k < 64; ++k) {
        t6 += th6r[t * 64 + k] * ls[k] + th6c[t * 64 + k] * mc[k];
        t7 += th7[t * 64 + k] * mz[k];
    }
    feat[t]      = 1.f / (1.f + expf(-t6));
    feat[64 + t] = 1.f / (1.f + expf(-t7));
    __syncthreads();
    if (t < 2) {
        float o = b8[t];
        for (int j = 0; j < 128; ++j) o += W8[t * 128 + j] * feat[j];
        out[t] = o;
    }
}

// ---------------------------------------------------------------------------
extern "C" void kernel_launch(void* const* d_in, const int* in_sizes, int n_in,
                              void* d_out, int out_size, void* d_ws, size_t ws_size,
                              hipStream_t stream) {
    (void)in_sizes; (void)n_in; (void)out_size; (void)ws_size;
    const float* A     = (const float*)d_in[0];
    const float* b     = (const float*)d_in[1];
    const float* c     = (const float*)d_in[2];
    const float* th1r  = (const float*)d_in[3];
    const float* th1c  = (const float*)d_in[4];
    const float* th2rr = (const float*)d_in[5];
    const float* th2rc = (const float*)d_in[6];
    const float* th2cr = (const float*)d_in[7];
    const float* th3rr = (const float*)d_in[8];
    const float* th3rc = (const float*)d_in[9];
    const float* th3cr = (const float*)d_in[10];
    const float* th4rr = (const float*)d_in[11];
    const float* th4rc = (const float*)d_in[12];
    const float* th4cr = (const float*)d_in[13];
    const float* th6r  = (const float*)d_in[14];
    const float* th6c  = (const float*)d_in[15];
    const float* th7   = (const float*)d_in[16];
    const float* W8    = (const float*)d_in[17];
    const float* b8    = (const float*)d_in[18];
    const int*   zp    = (const int*)d_in[19];

    float* wsf     = (float*)d_ws;
    float* srr_pos = wsf + 0;        // 8000
    float* srr_neg = wsf + 8000;     // 8000
    float* rowsum  = wsf + 16000;    // 64
    float* scal    = wsf + 16064;    // Sp, Sn
    float* wrc     = wsf + 16128;    // 8000
    float* fc      = wsf + 24128;    // 64
    float* fcraw   = wsf + 24192;    // 64
    float* t1r     = wsf + 24256;    // 64
    float* t1c     = wsf + 24320;    // 64
    float* evec    = wsf + 24384;    // 4*64
    float* edge_c  = wsf + 24640;    // 64
    float* vconst  = wsf + 24704;    // 64
    float* mu_c    = wsf + 24768;    // 64
    float* last_rs = wsf + 24832;    // 64
    _Float16* hbase   = (_Float16*)(wsf + 24896);
    _Float16* Fr16    = hbase;                 // 8000*64
    _Float16* th2rr16 = hbase + 512000;        // 64*64
    _Float16* muA     = hbase + 516096;        // 8000*64
    _Float16* muB     = hbase + 1028096;       // 8000*64
    float* out = (float*)d_out;

    k_zero<<<(16128 + 255) / 256, 256, 0, stream>>>(wsf, 16128);
    k_fc<<<1, 64, 0, stream>>>(c, fc, fcraw);
    k_rows<<<1000, 256, 0, stream>>>(A, b, fc, wrc, scal, Fr16, muA);
    k_gram<<<dim3(125, USPLIT), 64, 0, stream>>>(Fr16, srr_pos, srr_neg);
    k_prep<<<1, 64, 0, stream>>>(A, b, zp, th1r, th1c, th3rr, th3rc, th3cr,
                                 th4rr, th4rc, th4cr, th2rc, th2rr, scal, fcraw,
                                 t1r, t1c, evec, edge_c, vconst, mu_c, th2rr16);
    _Float16* cur = muA;
    _Float16* nxt = muB;
    for (int step = 0; step < 5; ++step) {
        k_row<<<125, 128, 0, stream>>>(cur, nxt, th2rr16, vconst, evec,
                                       srr_pos, srr_neg, wrc, rowsum);
        k_col<<<1, 64, 0, stream>>>(t1c, t1r, th2cr, th2rc, edge_c,
                                    rowsum, mu_c, vconst, last_rs);
        _Float16* tmp = cur; cur = nxt; nxt = tmp;
    }
    k_final<<<1, 64, 0, stream>>>(th6r, th6c, th7, W8, b8, zp, cur,
                                  last_rs, mu_c, out);
}
